// HybridGraphConv_44367012168180
// MI455X (gfx1250) — matrix operand order repaired and checked by reading, not compiled, not run on hardware
//
#include <hip/hip_runtime.h>
#include <hip/hip_bf16.h>

// ---------------------------------------------------------------------------
// HybridGraphConv for MI455X (gfx1250): bf16 WMMA GEMMs (pre-swizzled B
// fragments, b128 loads, fully unrolled compile-time K) + atomic edge scatter.
// ---------------------------------------------------------------------------

typedef __attribute__((ext_vector_type(16))) __bf16 v16bf;
typedef __attribute__((ext_vector_type(8)))  float  v8f;

#define CIN 128
#define CC  128
#define NH  4

// ---- helpers --------------------------------------------------------------

__device__ __forceinline__ unsigned short f2bf(float f) {
    unsigned u = __float_as_uint(f);
    unsigned r = (u + 0x7FFFu + ((u >> 16) & 1u)) >> 16;   // round-nearest-even
    return (unsigned short)r;
}

__device__ __forceinline__ unsigned enc_f32(float f) {
    unsigned u = __float_as_uint(f);
    return (u & 0x80000000u) ? ~u : (u | 0x80000000u);     // order-preserving
}
__device__ __forceinline__ float dec_f32(unsigned e) {
    return __uint_as_float((e & 0x80000000u) ? (e & 0x7FFFFFFFu) : ~e);
}

__device__ __forceinline__ float leaky(float v) { return v > 0.f ? v : 0.2f * v; }

// ---- small utility kernels ------------------------------------------------

__global__ void fill_u32_kernel(unsigned* __restrict__ p, unsigned v, size_t n) {
    size_t i = (size_t)blockIdx.x * blockDim.x + threadIdx.x;
    size_t stride = (size_t)gridDim.x * blockDim.x;
    for (; i < n; i += stride) p[i] = v;
}

// A-side (row-major) f32 -> bf16 conversion
__global__ void pack_bf16_kernel(const float* __restrict__ s, size_t total,
                                 unsigned short* __restrict__ d) {
    size_t i = (size_t)blockIdx.x * blockDim.x + threadIdx.x;
    size_t stride = (size_t)gridDim.x * blockDim.x;
    for (; i < total; i += stride) d[i] = f2bf(s[i]);
}

// B-side: swizzle f32 W[K,M] into B-fragment-native bf16 layout inside a
// packed matrix with Mtot total columns, at column offset coloff.
// Layout: dst[(((kt*Mtot + n)*2 + half)*16) + j] = W[kt*32 + half*16 + j][n]
// so one lane's 16 K-values for its column are 32 contiguous bytes.
__global__ void pack_wb_kernel(const float* __restrict__ s, int K, int M,
                               unsigned short* __restrict__ d, int Mtot, int coloff) {
    size_t total = (size_t)K * M;
    size_t i = (size_t)blockIdx.x * blockDim.x + threadIdx.x;
    size_t stride = (size_t)gridDim.x * blockDim.x;
    for (; i < total; i += stride) {
        int k = (int)(i / M), n = (int)(i % M);
        int kt = k >> 5, r = k & 31, half = r >> 4, j = r & 15;
        size_t o = (((size_t)kt * Mtot + coloff + n) * 2 + half) * 16 + j;
        d[o] = f2bf(s[i]);
    }
}

// ---- generic bf16 WMMA GEMM: C[nrows,M] (=/+=) A[nrows,KDIM] @ W[KDIM,M] --
// A row-major bf16; W pre-swizzled (pack_wb_kernel) with wcols total columns.
// block = 128 threads (4 waves). Each wave: one 16-row strip, 4x 16-col tiles.
// KDIM is compile-time: fully unrolled k-loop, no register rotation, loads of
// step i+1 can overlap WMMAs of step i.
// grid = (ceil(nrows/64), wcols/64)

union Frag16 { uint4 q[2]; v16bf v; };

__device__ __forceinline__ void load_a_frag(Frag16& a, const unsigned short* arow,
                                            int kk, int half) {
    a.q[0] = *reinterpret_cast<const uint4*>(arow + kk + half * 16);
    a.q[1] = *reinterpret_cast<const uint4*>(arow + kk + half * 16 + 8);
}
__device__ __forceinline__ void load_b_frag(Frag16& b, const unsigned short* W,
                                            int kt, int wcols, int n, int half) {
    const unsigned short* wp = W + ((size_t)kt * wcols + n) * 32 + half * 16;
    b.q[0] = *reinterpret_cast<const uint4*>(wp);
    b.q[1] = *reinterpret_cast<const uint4*>(wp + 8);
}

template <int KDIM>
__global__ void gemm_bf16_wmma(const unsigned short* __restrict__ A, int lda,
                               const unsigned short* __restrict__ W, int wcols,
                               float* __restrict__ C, int ldc,
                               int nrows, int accum) {
    const int wave = threadIdx.x >> 5;
    const int lane = threadIdx.x & 31;
    const int row0 = blockIdx.x * 64 + wave * 16;
    if (row0 >= nrows) return;                    // wave-uniform; EXEC stays full
    const int col0 = blockIdx.y * 64;
    const int m    = lane & 15;
    const int half = lane >> 4;

    v8f acc[4];
#pragma unroll
    for (int i = 0; i < 4; i++)
#pragma unroll
        for (int j = 0; j < 8; j++) acc[i][j] = 0.f;

    const unsigned short* arow = A + (size_t)(row0 + m) * lda;

#pragma unroll
    for (int kk = 0; kk < KDIM; kk += 32) {
        Frag16 a, b[4];
        load_a_frag(a, arow, kk, half);
#pragma unroll
        for (int ct = 0; ct < 4; ct++)
            load_b_frag(b[ct], W, kk >> 5, wcols, col0 + ct * 16 + m, half);
#pragma unroll
        for (int ct = 0; ct < 4; ct++)
            acc[ct] = __builtin_amdgcn_wmma_f32_16x16x32_bf16(
                false, a.v, false, b[ct].v, (short)0, acc[ct], false, false);
    }
#pragma unroll
    for (int ct = 0; ct < 4; ct++)
#pragma unroll
        for (int r = 0; r < 8; r++) {
            const int row = row0 + half * 8 + r;
            const int col = col0 + ct * 16 + m;
            float* cp = C + (size_t)row * ldc + col;
            float v = acc[ct][r];
            if (accum) v += *cp;
            *cp = v;
        }
}

// ---- GAT attention logits per node ---------------------------------------
// block 128 = 4 heads x 32 lanes, one node per block
__global__ void a_sd_kernel(const float* __restrict__ hhx,
                            const float* __restrict__ asrc,
                            const float* __restrict__ adst,
                            float* __restrict__ a_s, float* __restrict__ a_d) {
    const int n = blockIdx.x;
    const int h = threadIdx.x >> 5, l = threadIdx.x & 31;
    const float* hg = hhx + (size_t)n * 768 + 128 + h * 128;
    float ss = 0.f, sd = 0.f;
#pragma unroll
    for (int c = l; c < 128; c += 32) {
        float v = hg[c];
        ss += v * asrc[h * 128 + c];
        sd += v * adst[h * 128 + c];
    }
#pragma unroll
    for (int off = 16; off; off >>= 1) {
        ss += __shfl_down(ss, off, 32);
        sd += __shfl_down(sd, off, 32);
    }
    if (l == 0) { a_s[n * 4 + h] = ss; a_d[n * 4 + h] = sd; }
}

// ---- edge pass A: in-degree + segment max of logits -----------------------
__global__ void edge_pass_a(const long long* __restrict__ ei, long long E,
                            const float* __restrict__ a_s, const float* __restrict__ a_d,
                            float* __restrict__ deg, unsigned* __restrict__ m_enc) {
    long long e = (long long)blockIdx.x * blockDim.x + threadIdx.x;
    if (e >= E) return;
    const int s = (int)ei[e], d = (int)ei[E + e];
    atomicAdd(deg + d, 1.0f);
#pragma unroll
    for (int h = 0; h < 4; h++) {
        float ev = leaky(a_s[s * 4 + h] + a_d[d * 4 + h]);
        atomicMax(m_enc + (size_t)d * 4 + h, enc_f32(ev));
    }
}

// ---- node mid: fold self-loop into max; dinv = rsqrt(deg+1) ---------------
__global__ void node_mid(const float* __restrict__ deg, const unsigned* __restrict__ m_enc,
                         const float* __restrict__ a_s, const float* __restrict__ a_d,
                         float* __restrict__ m_fin, float* __restrict__ dinv, int N) {
    int t = blockIdx.x * blockDim.x + threadIdx.x;
    if (t < N * 4) {
        int n = t >> 2, h = t & 3;
        float es = leaky(a_s[n * 4 + h] + a_d[n * 4 + h]);   // self-loop logit
        m_fin[t] = fmaxf(dec_f32(m_enc[t]), es);
    }
    if (t < N) dinv[t] = rsqrtf(deg[t] + 1.0f);              // +1 self loop (GCN)
}

// ---- edge pass B (heavy gather/scatter): one wave per edge ----------------
__global__ void edge_pass_b(const long long* __restrict__ ei, long long E,
                            const float* __restrict__ hhx, const float* __restrict__ x,
                            const float* __restrict__ a_s, const float* __restrict__ a_d,
                            const float* __restrict__ m_fin, const float* __restrict__ dinv,
                            float* __restrict__ gcn_acc, float* __restrict__ sage_acc,
                            float* __restrict__ gat_acc, float* __restrict__ denom) {
    const int lane = threadIdx.x & 31;
    long long e = (long long)blockIdx.x * (blockDim.x >> 5) + (threadIdx.x >> 5);
    if (e >= E) return;
    const int s = (int)ei[e], d = (int)ei[E + e];

    // prefetch a future edge's source rows (gfx1250 global_prefetch_b8)
    long long ep = e + 256;
    if (ep < E) {
        int sp = (int)ei[ep];
        __builtin_prefetch(hhx + (size_t)sp * 768 + lane * 24, 0, 1);
        __builtin_prefetch(x + (size_t)sp * 128 + lane * 4, 0, 1);
    }

    const float  dv   = dinv[s];
    const float* hrow = hhx + (size_t)s * 768;        // h
    const float* xrow = x + (size_t)s * 128;
#pragma unroll
    for (int it = 0; it < 4; it++) {
        int c = it * 32 + lane;
        atomicAdd(gcn_acc + (size_t)d * 128 + c, hrow[c] * dv);
        atomicAdd(sage_acc + (size_t)d * 128 + c, xrow[c]);
    }
    float ex0, ex1, ex2, ex3;
    {
        float e0 = leaky(a_s[s * 4 + 0] + a_d[d * 4 + 0]);
        float e1 = leaky(a_s[s * 4 + 1] + a_d[d * 4 + 1]);
        float e2 = leaky(a_s[s * 4 + 2] + a_d[d * 4 + 2]);
        float e3 = leaky(a_s[s * 4 + 3] + a_d[d * 4 + 3]);
        ex0 = __expf(e0 - m_fin[(size_t)d * 4 + 0]);
        ex1 = __expf(e1 - m_fin[(size_t)d * 4 + 1]);
        ex2 = __expf(e2 - m_fin[(size_t)d * 4 + 2]);
        ex3 = __expf(e3 - m_fin[(size_t)d * 4 + 3]);
    }
    if (lane < 4) {
        float v = lane == 0 ? ex0 : lane == 1 ? ex1 : lane == 2 ? ex2 : ex3;
        atomicAdd(denom + (size_t)d * 4 + lane, v);
    }
    const float* hgrow = hrow + 128;                  // hg (4 heads x 128)
#pragma unroll
    for (int hh = 0; hh < 4; hh++) {
        float exh = hh == 0 ? ex0 : hh == 1 ? ex1 : hh == 2 ? ex2 : ex3;
#pragma unroll
        for (int it = 0; it < 4; it++) {
            int c = hh * 128 + it * 32 + lane;
            atomicAdd(gat_acc + (size_t)d * 512 + c, hgrow[c] * exh);
        }
    }
}

// ---- epilogue 1: finish GAT softmax + relu(+b) -> bf16; SAGE mean -> bf16 -
__global__ void epi1_kernel(const float* __restrict__ hhx, const float* __restrict__ gat_acc,
                            const float* __restrict__ a_s, const float* __restrict__ a_d,
                            const float* __restrict__ m_fin, const float* __restrict__ denom,
                            const float* __restrict__ deg, const float* __restrict__ sage_acc,
                            const float* __restrict__ gat_b,
                            unsigned short* __restrict__ gat_pre,
                            unsigned short* __restrict__ nbrm) {
    const int n = blockIdx.x, t = threadIdx.x;
    const size_t n4 = (size_t)n * 4, n512 = (size_t)n * 512;
    const float* hgrow = hhx + (size_t)n * 768 + 128;
#pragma unroll
    for (int h = 0; h < 4; h++) {
        float ev = leaky(a_s[n4 + h] + a_d[n4 + h]);
        float es = __expf(ev - m_fin[n4 + h]);            // self-loop weight
        float dt = denom[n4 + h] + es;
        int c = h * 128 + t;
        float raw = (gat_acc[n512 + c] + hgrow[c] * es) / dt;
        gat_pre[n512 + c] = f2bf(fmaxf(raw + gat_b[c], 0.f));
    }
    float nm = sage_acc[(size_t)n * 128 + t] / fmaxf(deg[n], 1.0f);
    nbrm[(size_t)n * 128 + t] = f2bf(nm);
}

// ---- merge three branches, learned softmax fuse, LayerNorm -> bf16 --------
__global__ void merge_ln_kernel(const float* __restrict__ hhx, const float* __restrict__ gcn_acc,
                                const float* __restrict__ dinv, const float* __restrict__ gcn_b,
                                const float* __restrict__ outg, const float* __restrict__ gat_tb,
                                const float* __restrict__ outs, const float* __restrict__ sage_bl,
                                const float* __restrict__ attn_w,
                                const float* __restrict__ ln_g, const float* __restrict__ ln_b,
                                unsigned short* __restrict__ mergedb) {
    __shared__ float sh[8];
    const int n = blockIdx.x, t = threadIdx.x;
    const size_t o = (size_t)n * 128 + t;
    const float dv = dinv[n];
    float hval = hhx[(size_t)n * 768 + t];
    float gcn = fmaxf(dv * (gcn_acc[o] + hval * dv) + gcn_b[t], 0.f);
    float gat = outg[o] + gat_tb[t];
    float xr  = hhx[(size_t)n * 768 + 640 + t];
    float sage = fmaxf(outs[o] + sage_bl[t] + xr, 0.f);

    float w0 = attn_w[0], w1 = attn_w[1], w2 = attn_w[2];
    float mx = fmaxf(w0, fmaxf(w1, w2));
    float e0 = __expf(w0 - mx), e1 = __expf(w1 - mx), e2 = __expf(w2 - mx);
    float inv = 1.f / (e0 + e1 + e2);
    float v = (e0 * gcn + e1 * gat + e2 * sage) * inv;

    float s1 = v, s2 = v * v;
#pragma unroll
    for (int off = 16; off; off >>= 1) {
        s1 += __shfl_down(s1, off, 32);
        s2 += __shfl_down(s2, off, 32);
    }
    const int wv = t >> 5, l = t & 31;
    if (l == 0) { sh[wv] = s1; sh[4 + wv] = s2; }
    __syncthreads();
    s1 = sh[0] + sh[1] + sh[2] + sh[3];
    s2 = sh[4] + sh[5] + sh[6] + sh[7];
    float mu = s1 * (1.f / 128.f);
    float var = s2 * (1.f / 128.f) - mu * mu;
    float nv = (v - mu) * rsqrtf(var + 1e-5f) * ln_g[t] + ln_b[t];
    mergedb[o] = f2bf(nv);
}

// ---- final: out = relu(outf + fus_b + x) ----------------------------------
__global__ void final_kernel(const float* __restrict__ outf, const float* __restrict__ fus_b,
                             const float* __restrict__ x, float* __restrict__ out, size_t total) {
    size_t i = (size_t)blockIdx.x * blockDim.x + threadIdx.x;
    if (i >= total) return;
    int c = (int)(i & 127);
    out[i] = fmaxf(outf[i] + fus_b[c] + x[i], 0.f);
}

// ---------------------------------------------------------------------------

static inline unsigned cdiv(long long a, long long b) { return (unsigned)((a + b - 1) / b); }

extern "C" void kernel_launch(void* const* d_in, const int* in_sizes, int n_in,
                              void* d_out, int out_size, void* d_ws, size_t ws_size,
                              hipStream_t stream) {
    (void)n_in; (void)out_size; (void)ws_size;
    const float* x        = (const float*)d_in[0];
    const float* gcn_w    = (const float*)d_in[1];
    const float* gcn_b    = (const float*)d_in[2];
    const float* gat_w    = (const float*)d_in[3];
    const float* gat_asrc = (const float*)d_in[4];
    const float* gat_adst = (const float*)d_in[5];
    const float* gat_b    = (const float*)d_in[6];
    const float* gat_tw   = (const float*)d_in[7];
    const float* gat_tb   = (const float*)d_in[8];
    const float* sage_wl  = (const float*)d_in[9];
    const float* sage_bl  = (const float*)d_in[10];
    const float* sage_wr  = (const float*)d_in[11];
    const float* attn_w   = (const float*)d_in[12];
    const float* ln_g     = (const float*)d_in[13];
    const float* ln_b     = (const float*)d_in[14];
    const float* fus_w    = (const float*)d_in[15];
    const float* fus_b    = (const float*)d_in[16];
    const long long* ei   = (const long long*)d_in[17];

    const int       N = in_sizes[0] / CIN;     // 20000
    const long long E = in_sizes[17] / 2;      // 320000

    // ---- workspace carve-out (256B aligned) ----
    char* ws = (char*)d_ws;
    size_t off = 0;
    auto alloc = [&](size_t bytes) -> void* {
        void* p = ws + off;
        off += (bytes + 255) & ~(size_t)255;
        return p;
    };
    unsigned short* xb      = (unsigned short*)alloc((size_t)N * 128 * 2);
    unsigned short* w1      = (unsigned short*)alloc((size_t)128 * 768 * 2);
    unsigned short* w2g     = (unsigned short*)alloc((size_t)512 * 128 * 2);
    unsigned short* w2s     = (unsigned short*)alloc((size_t)128 * 128 * 2);
    unsigned short* wf      = (unsigned short*)alloc((size_t)256 * 128 * 2);
    float*          hhx     = (float*)alloc((size_t)N * 768 * 4);   // [h | hg | xr]
    float*          a_s     = (float*)alloc((size_t)N * 4 * 4);
    float*          a_d     = (float*)alloc((size_t)N * 4 * 4);
    unsigned*       m_enc   = (unsigned*)alloc((size_t)N * 4 * 4);
    float*          m_fin   = (float*)alloc((size_t)N * 4 * 4);
    float*          dinv    = (float*)alloc((size_t)N * 4);
    float*          accblk  = (float*)alloc((size_t)N * 773 * 4);   // deg|denom|gcn|sage|gat
    float*          deg     = accblk;
    float*          denom   = accblk + (size_t)N;
    float*          gcn_acc = denom + (size_t)N * 4;
    float*          sage_acc= gcn_acc + (size_t)N * 128;
    float*          gat_acc = sage_acc + (size_t)N * 128;
    unsigned short* gat_pre = (unsigned short*)alloc((size_t)N * 512 * 2);
    unsigned short* nbrm    = (unsigned short*)alloc((size_t)N * 128 * 2);
    float*          outg    = (float*)alloc((size_t)N * 128 * 4);
    float*          outs    = (float*)alloc((size_t)N * 128 * 4);
    unsigned short* mergedb = (unsigned short*)alloc((size_t)N * 128 * 2);
    float*          outf    = (float*)alloc((size_t)N * 128 * 4);
    float*          outp    = (float*)d_out;

    const unsigned ENC_NEG_INF = 0x007FFFFFu;   // enc_f32(-inf)

    // 1) zero accumulators, init segment-max
    {
        size_t cnt = (size_t)N * 773;
        fill_u32_kernel<<<cdiv(cnt, 256), 256, 0, stream>>>((unsigned*)accblk, 0u, cnt);
        fill_u32_kernel<<<cdiv((size_t)N * 4, 256), 256, 0, stream>>>(m_enc, ENC_NEG_INF, (size_t)N * 4);
    }
    // 2) bf16 conversions / B-side weight swizzling
    pack_bf16_kernel<<<cdiv((size_t)N * 128, 256), 256, 0, stream>>>(x, (size_t)N * 128, xb);
    pack_wb_kernel<<<cdiv(128 * 128, 256), 256, 0, stream>>>(gcn_w, 128, 128, w1, 768, 0);
    pack_wb_kernel<<<cdiv(128 * 512, 256), 256, 0, stream>>>(gat_w, 128, 512, w1, 768, 128);
    pack_wb_kernel<<<cdiv(128 * 128, 256), 256, 0, stream>>>(sage_wr, 128, 128, w1, 768, 640);
    pack_wb_kernel<<<cdiv(512 * 128, 256), 256, 0, stream>>>(gat_tw, 512, 128, w2g, 128, 0);
    pack_wb_kernel<<<cdiv(128 * 128, 256), 256, 0, stream>>>(sage_wl, 128, 128, w2s, 128, 0);
    pack_wb_kernel<<<cdiv(256 * 128, 256), 256, 0, stream>>>(fus_w, 256, 128, wf, 128, 0);

    // 3) GEMM1: [h | hg | xr] = xb @ w1   ([N,128] x [128,768])
    gemm_bf16_wmma<128><<<dim3(cdiv(N, 64), 12), 128, 0, stream>>>(xb, 128, w1, 768, hhx, 768, N, 0);

    // 4) attention logits
    a_sd_kernel<<<N, 128, 0, stream>>>(hhx, gat_asrc, gat_adst, a_s, a_d);

    // 5) edge pass A: degree + segment max
    edge_pass_a<<<cdiv(E, 256), 256, 0, stream>>>(ei, E, a_s, a_d, deg, m_enc);

    // 6) node mid: fold self-loops, dinv
    node_mid<<<cdiv((long long)N * 4, 256), 256, 0, stream>>>(deg, m_enc, a_s, a_d, m_fin, dinv, N);

    // 7) edge pass B: heavy gather/scatter (wave per edge, 8 waves/block)
    edge_pass_b<<<cdiv(E, 8), 256, 0, stream>>>(ei, E, hhx, x, a_s, a_d, m_fin, dinv,
                                                gcn_acc, sage_acc, gat_acc, denom);

    // 8) epilogue 1: finish GAT softmax, SAGE mean
    epi1_kernel<<<N, 128, 0, stream>>>(hhx, gat_acc, a_s, a_d, m_fin, denom, deg, sage_acc,
                                       gat_b, gat_pre, nbrm);

    // 9) GEMM2a: gat_t = gat_pre @ gat_tw   ([N,512] x [512,128])
    gemm_bf16_wmma<512><<<dim3(cdiv(N, 64), 2), 128, 0, stream>>>(gat_pre, 512, w2g, 128, outg, 128, N, 0);
    // 10) GEMM2b: sage_mm = nbr_mean @ sage_wl
    gemm_bf16_wmma<128><<<dim3(cdiv(N, 64), 2), 128, 0, stream>>>(nbrm, 128, w2s, 128, outs, 128, N, 0);

    // 11) merge + softmax fuse + LayerNorm -> bf16
    merge_ln_kernel<<<N, 128, 0, stream>>>(hhx, gcn_acc, dinv, gcn_b, outg, gat_tb, outs,
                                           sage_bl, attn_w, ln_g, ln_b, mergedb);

    // 12) fusion GEMM split: outf = merged @ fus_w[:128] + x @ fus_w[128:]
    //     (wf + 128*128 == k-tiles 4..7 of the packed [256,128] weight)
    gemm_bf16_wmma<128><<<dim3(cdiv(N, 64), 2), 128, 0, stream>>>(mergedb, 128, wf, 128, outf, 128, N, 0);
    gemm_bf16_wmma<128><<<dim3(cdiv(N, 64), 2), 128, 0, stream>>>(xb, 128, wf + 128 * 128, 128, outf, 128, N, 1);

    // 13) bias + residual + relu
    final_kernel<<<cdiv((size_t)N * 128, 256), 256, 0, stream>>>(outf, fus_b, x, outp, (size_t)N * 128);
}
